// MultiHeadAttention_47845935677543
// MI455X (gfx1250) — matrix operand order repaired and checked by reading, not compiled, not run on hardware
//
#include <hip/hip_runtime.h>
#include <stdint.h>

// ---------------------------------------------------------------------------
// MHA: B=4, S=2048, E=1024, H=16, DH=64. All GEMMs via v_wmma_f32_16x16x32_bf16.
// 2x2 register tiling everywhere; attention stages K/V tiles into LDS with
// CDNA5 async global->LDS DMA (ASYNCcnt), double-buffered, shared by the
// 8 waves of a block (8x cut in K/V L2 traffic).
// ---------------------------------------------------------------------------

#define Bn 4
#define Sn 2048
#define En 1024
#define Hn 16
#define DHn 64

typedef __attribute__((ext_vector_type(8)))  float  v8f;
typedef __attribute__((ext_vector_type(16))) __bf16 v16bf;
typedef __attribute__((address_space(3))) unsigned short lds_us;

// one-shot RNE convert (off critical path)
__device__ __forceinline__ unsigned short f2bf(float f) {
    union { float f; unsigned u; } x; x.f = f;
    unsigned u = x.u;
    unsigned r = (u + 0x7FFFu + ((u >> 16) & 1u)) >> 16;  // RNE
    return (unsigned short)r;
}
// hot-path truncating convert: single shift
__device__ __forceinline__ unsigned short f2bf_t(float f) {
    return (unsigned short)(__float_as_uint(f) >> 16);
}
// pack two truncated bf16 into one dword: single v_perm_b32
__device__ __forceinline__ unsigned pk2t(float lo, float hi) {
    return __builtin_amdgcn_perm(__float_as_uint(hi), __float_as_uint(lo), 0x07060302u);
}
// 16 contiguous bf16 (32B) -> B-operand fragment (global or LDS)
__device__ __forceinline__ v16bf ld16(const unsigned short* p) {
    union { uint4 q[2]; v16bf v; } u;
    u.q[0] = ((const uint4*)p)[0];
    u.q[1] = ((const uint4*)p)[1];
    return u.v;
}
// two runs of 8 contiguous bf16 (16B each) -> A-operand fragment
__device__ __forceinline__ v16bf ld8x2(const unsigned short* p0, const unsigned short* p1) {
    union { uint4 q[2]; v16bf v; } u;
    u.q[0] = *(const uint4*)p0;
    u.q[1] = *(const uint4*)p1;
    return u.v;
}
__device__ __forceinline__ v8f wmma_bf(v16bf a, v16bf b, v8f c) {
    return __builtin_amdgcn_wmma_f32_16x16x32_bf16(false, a, false, b, (short)0, c, false, false);
}
// 32-bit LDS offset of a generic pointer into a __shared__ array
__device__ __forceinline__ unsigned lds_off(const unsigned short* p) {
    return (unsigned)(uintptr_t)(lds_us*)p;
}
// async 16B global -> LDS copy (per-lane), tracked by ASYNCcnt
__device__ __forceinline__ void async_cp16(unsigned lds_addr, const unsigned short* gsrc) {
    asm volatile("global_load_async_to_lds_b128 %0, %1, off"
                 :: "v"(lds_addr), "v"(gsrc) : "memory");
}

// ---------------------------------------------------------------------------
// Kernel 0: one-shot xs (f32) -> Xbf (bf16), 8 elements/thread, v_perm packing.
// ---------------------------------------------------------------------------
__global__ __launch_bounds__(256) void xs2bf(
        const float* __restrict__ xs, unsigned int* __restrict__ Xbf) {
    int t = blockIdx.x * 256 + threadIdx.x;       // 1,048,576 threads
    const float* p = xs + (size_t)t * 8;
    float4 a = *(const float4*)p;
    float4 b = *(const float4*)(p + 4);
    uint4 o;
    o.x = pk2t(a.x, a.y);
    o.y = pk2t(a.z, a.w);
    o.z = pk2t(b.x, b.y);
    o.w = pk2t(b.z, b.w);
    ((uint4*)Xbf)[t] = o;
}

// ---------------------------------------------------------------------------
// Kernel 1: transpose weights to bf16, K-dim contiguous.
// Wt[w][h][d][e] = W{q,k,v}[h][e][d];  WoT[o][e] = Wo[e][o]
// ---------------------------------------------------------------------------
__global__ __launch_bounds__(256) void prep_weights(
        const float* __restrict__ Wq, const float* __restrict__ Wk,
        const float* __restrict__ Wv, const float* __restrict__ Wo,
        unsigned short* __restrict__ Wt, unsigned short* __restrict__ WoT) {
    int idx = blockIdx.x * 256 + threadIdx.x;
    const int NWT = 3 * Hn * DHn * En;          // 3,145,728
    if (idx < NWT) {
        int e = idx & (En - 1);
        int d = (idx >> 10) & (DHn - 1);
        int h = (idx >> 16) & (Hn - 1);
        int w = idx >> 20;
        const float* W = (w == 0) ? Wq : ((w == 1) ? Wk : Wv);
        Wt[idx] = f2bf(W[((size_t)h * En + e) * DHn + d]);
    } else {
        int k = idx - NWT;
        if (k < En * En) {
            int e = k & (En - 1);
            int o = k >> 10;
            WoT[k] = f2bf(Wo[(size_t)e * En + o]);
        }
    }
}

// ---------------------------------------------------------------------------
// Kernel 2: QKV projection, 32x32 tile per wave (2x2 WMMA tiling), K-loop=32.
// Q scaled by 1/sqrt(DH). Outputs: Qbf/Kbf [b][h][s][d], Vt [b][h][d][s].
// ---------------------------------------------------------------------------
__global__ __launch_bounds__(256) void qkv_proj(
        const unsigned short* __restrict__ Xbf, const unsigned short* __restrict__ Wt,
        const float* __restrict__ bq, const float* __restrict__ bk,
        const float* __restrict__ bv,
        unsigned short* __restrict__ Qbf, unsigned short* __restrict__ Kbf,
        unsigned short* __restrict__ Vt) {
    int lane = threadIdx.x & 31, ln = lane & 15, half = lane >> 4;
    int wid = blockIdx.x * 8 + (threadIdx.x >> 5);
    int st2 = wid & 255;                // 32-row tile over B*S (256 of them)
    int r   = wid >> 8;
    int dcp = r & 1;                    // 32-col chunk of DH (2 of them)
    int h   = (r >> 1) & 15;
    int w   = r >> 5;                   // 0=Q,1=K,2=V
    int rowbase = st2 * 32;

    const unsigned short* arow0 = Xbf + (size_t)(rowbase + ln) * En;
    const unsigned short* arow1 = arow0 + (size_t)16 * En;
    const unsigned short* wrow0 = Wt + (size_t)(((w * Hn + h) * DHn) + dcp * 32 + ln) * En;
    const unsigned short* wrow1 = wrow0 + (size_t)16 * En;

    v8f c00 = {0.f,0.f,0.f,0.f,0.f,0.f,0.f,0.f};
    v8f c01 = c00, c10 = c00, c11 = c00;
#pragma unroll 2
    for (int kk = 0; kk < 32; ++kk) {
        v16bf A0 = ld8x2(arow0 + kk * 32 + half * 16, arow0 + kk * 32 + half * 16 + 8);
        v16bf A1 = ld8x2(arow1 + kk * 32 + half * 16, arow1 + kk * 32 + half * 16 + 8);
        v16bf B0 = ld16(wrow0 + kk * 32 + half * 16);
        v16bf B1 = ld16(wrow1 + kk * 32 + half * 16);
        c00 = wmma_bf(A0, B0, c00);
        c01 = wmma_bf(A0, B1, c01);
        c10 = wmma_bf(A1, B0, c10);
        c11 = wmma_bf(A1, B1, c11);
    }
    const float* bptr = (w == 0) ? bq : ((w == 1) ? bk : bv);
    v8f cc[2][2] = {{c00, c01}, {c10, c11}};
#pragma unroll
    for (int rt = 0; rt < 2; ++rt) {
#pragma unroll
        for (int ct = 0; ct < 2; ++ct) {
            int col = dcp * 32 + ct * 16 + ln;
            float bias = bptr[h * DHn + col];
#pragma unroll
            for (int j = 0; j < 8; ++j) {
                int row = rowbase + rt * 16 + j + 8 * half;
                int b = row >> 11, s = row & (Sn - 1);
                float v = cc[rt][ct][j] + bias;
                size_t bh = (size_t)(b * Hn + h);
                if (w == 0)      Qbf[(bh * Sn + s) * DHn + col] = f2bf_t(v * 0.125f);
                else if (w == 1) Kbf[(bh * Sn + s) * DHn + col] = f2bf_t(v);
                else             Vt [(bh * DHn + col) * Sn + s] = f2bf_t(v);
            }
        }
    }
}

// ---------------------------------------------------------------------------
// Kernel 3: flash attention. Block = 8 waves sharing one (b,h); each wave owns
// a 32-row query tile. K/V 32-column tiles are staged into LDS once per block
// via async global->LDS DMA (double-buffered, ASYNCcnt), then consumed by all
// 8 waves: 16 WMMA per t-step per wave against LDS-resident K/V.
// ---------------------------------------------------------------------------
__global__ __launch_bounds__(256) void attn(
        const unsigned short* __restrict__ Qbf, const unsigned short* __restrict__ Kbf,
        const unsigned short* __restrict__ Vt,  unsigned short* __restrict__ Ctx) {
    __shared__ __align__(16) unsigned short Kst[2 * 32 * 64];   //  8 KB: 2 x [t=32][d=64]
    __shared__ __align__(16) unsigned short Vst[2 * 64 * 32];   //  8 KB: 2 x [d=64][t=32]
    __shared__ __align__(16) unsigned short Pls[8 * 1024];      // 16 KB: 8 waves x 32x32

    int tid  = threadIdx.x;
    int lane = tid & 31, ln = lane & 15, half = lane >> 4;
    int wlocal = tid >> 5;
    unsigned short* lds = Pls + wlocal * 1024;

    int wid = blockIdx.x * 8 + wlocal;          // 4096 waves total
    int stp = wid & 63;                         // 32-row S tile
    int bh  = wid >> 6;                         // b*H + h (uniform per block)
    int rowbase = stp * 32;

    const unsigned short* Qb = Qbf + (size_t)bh * Sn * DHn;
    const unsigned short* Kb = Kbf + (size_t)bh * Sn * DHn;
    const unsigned short* Vb = Vt  + (size_t)bh * DHn * Sn;

    unsigned kst0 = lds_off(Kst), vst0 = lds_off(Vst);

    // Q A-operands (loop invariant): 2 row tiles x 2 K-fragments
    v16bf aq[2][2];
#pragma unroll
    for (int rt = 0; rt < 2; ++rt) {
        const unsigned short* qr = Qb + (size_t)(rowbase + rt * 16 + ln) * DHn;
        aq[rt][0] = ld8x2(qr + half * 16,      qr + 8 + half * 16);
        aq[rt][1] = ld8x2(qr + 32 + half * 16, qr + 40 + half * 16);
    }

    float m[2][8], l[2][8];
    v8f acc[2][4];
#pragma unroll
    for (int rt = 0; rt < 2; ++rt) {
#pragma unroll
        for (int j = 0; j < 8; ++j) { m[rt][j] = -3.0e38f; l[rt][j] = 0.f; }
#pragma unroll
        for (int n = 0; n < 4; ++n) acc[rt][n] = (v8f){0.f,0.f,0.f,0.f,0.f,0.f,0.f,0.f};
    }

    // ---- async stage issue: K tile is 4KB contiguous; V tile 64 rows x 64B
    auto issue_stage = [&](int tt, int buf) {
        int tb = tt * 32;
        async_cp16(kst0 + buf * 4096 + tid * 16, Kb + (size_t)tb * DHn + tid * 8);
        async_cp16(vst0 + buf * 4096 + tid * 16,
                   Vb + (size_t)(tid >> 2) * Sn + tb + (tid & 3) * 8);
    };

    issue_stage(0, 0);
    for (int tt = 0; tt < Sn / 32; ++tt) {
        int buf = tt & 1;
        __syncthreads();                         // prev compute done -> back buffer free
        if (tt + 1 < Sn / 32) {
            issue_stage(tt + 1, buf ^ 1);        // overlap DMA with compute below
            asm volatile("s_wait_asynccnt 0x2" ::: "memory");   // stage tt landed
        } else {
            asm volatile("s_wait_asynccnt 0x0" ::: "memory");
        }
        __syncthreads();                         // stage tt visible block-wide

        const unsigned short* Kl = Kst + buf * 2048;   // [t=32][d=64]
        const unsigned short* Vl = Vst + buf * 2048;   // [d=64][t=32]

        // K B-operands from LDS: 2 col tiles x 2 K-fragments, shared by row tiles
        v16bf bk00 = ld16(Kl + ln * 64 + half * 16);
        v16bf bk01 = ld16(Kl + ln * 64 + 32 + half * 16);
        v16bf bk10 = ld16(Kl + (16 + ln) * 64 + half * 16);
        v16bf bk11 = ld16(Kl + (16 + ln) * 64 + 32 + half * 16);

        v16bf ap[2];
#pragma unroll
        for (int rt = 0; rt < 2; ++rt) {
            v8f c0 = {0.f,0.f,0.f,0.f,0.f,0.f,0.f,0.f};
            v8f c1 = c0;
            c0 = wmma_bf(aq[rt][0], bk00, c0);
            c0 = wmma_bf(aq[rt][1], bk01, c0);
            c1 = wmma_bf(aq[rt][0], bk10, c1);
            c1 = wmma_bf(aq[rt][1], bk11, c1);
            // online softmax; C-layout: lane = col (ln), rows j+8*half per VGPR j
#pragma unroll
            for (int j = 0; j < 8; ++j) {
                float mr = fmaxf(c0[j], c1[j]);
                mr = fmaxf(mr, __shfl_xor(mr, 1, 32));
                mr = fmaxf(mr, __shfl_xor(mr, 2, 32));
                mr = fmaxf(mr, __shfl_xor(mr, 4, 32));
                mr = fmaxf(mr, __shfl_xor(mr, 8, 32));
                float mnew = fmaxf(m[rt][j], mr);
                float corr = __expf(m[rt][j] - mnew);
                m[rt][j] = mnew;
                float p0 = __expf(c0[j] - mnew);
                float p1 = __expf(c1[j] - mnew);
                float rs = p0 + p1;
                rs += __shfl_xor(rs, 1, 32);
                rs += __shfl_xor(rs, 2, 32);
                rs += __shfl_xor(rs, 4, 32);
                rs += __shfl_xor(rs, 8, 32);
                l[rt][j] = l[rt][j] * corr + rs;
                acc[rt][0][j] *= corr; acc[rt][1][j] *= corr;
                acc[rt][2][j] *= corr; acc[rt][3][j] *= corr;
                int rbase = (rt * 16 + j + 8 * half) * 32;
                lds[rbase + ln]      = f2bf_t(p0);
                lds[rbase + 16 + ln] = f2bf_t(p1);
            }
            // reload P in A-layout (LDS bounce = transpose); same-wave DS in-order
            ap[rt] = ld8x2(lds + (rt * 16 + ln) * 32 + half * 16,
                           lds + (rt * 16 + ln) * 32 + half * 16 + 8);
        }
        // ctx += P @ V ; each LDS V fragment feeds both row tiles
#pragma unroll
        for (int n = 0; n < 4; ++n) {
            v16bf bv = ld16(Vl + (n * 16 + ln) * 32 + half * 16);
            acc[0][n] = wmma_bf(ap[0], bv, acc[0][n]);
            acc[1][n] = wmma_bf(ap[1], bv, acc[1][n]);
        }
    }
    // normalize and write concat layout Ctx[b][s][h*64+d] (bf16)
    int b = bh >> 4, h = bh & 15;
#pragma unroll
    for (int rt = 0; rt < 2; ++rt) {
#pragma unroll
        for (int j = 0; j < 8; ++j) {
            float inv = 1.0f / l[rt][j];
            int s = rowbase + rt * 16 + j + 8 * half;
            size_t rb = ((size_t)b * Sn + s) * En + h * DHn;
#pragma unroll
            for (int n = 0; n < 4; ++n)
                Ctx[rb + n * 16 + ln] = f2bf_t(acc[rt][n][j] * inv);
        }
    }
}

// ---------------------------------------------------------------------------
// Kernel 4: output projection, 32x32 tile per wave. out = Ctx @ Wo + bo (f32).
// ---------------------------------------------------------------------------
__global__ __launch_bounds__(256) void out_proj(
        const unsigned short* __restrict__ Ctx, const unsigned short* __restrict__ WoT,
        const float* __restrict__ bo, float* __restrict__ out) {
    int lane = threadIdx.x & 31, ln = lane & 15, half = lane >> 4;
    int wid = blockIdx.x * 8 + (threadIdx.x >> 5);
    int st2 = wid & 255;                // 32-row tile
    int ocp = wid >> 8;                 // 0..31 32-col tile
    int rowbase = st2 * 32;

    const unsigned short* arow0 = Ctx + (size_t)(rowbase + ln) * En;
    const unsigned short* arow1 = arow0 + (size_t)16 * En;
    const unsigned short* wrow0 = WoT + (size_t)(ocp * 32 + ln) * En;
    const unsigned short* wrow1 = wrow0 + (size_t)16 * En;

    v8f c00 = {0.f,0.f,0.f,0.f,0.f,0.f,0.f,0.f};
    v8f c01 = c00, c10 = c00, c11 = c00;
#pragma unroll 2
    for (int kk = 0; kk < 32; ++kk) {
        v16bf A0 = ld8x2(arow0 + kk * 32 + half * 16, arow0 + kk * 32 + half * 16 + 8);
        v16bf A1 = ld8x2(arow1 + kk * 32 + half * 16, arow1 + kk * 32 + half * 16 + 8);
        v16bf B0 = ld16(wrow0 + kk * 32 + half * 16);
        v16bf B1 = ld16(wrow1 + kk * 32 + half * 16);
        c00 = wmma_bf(A0, B0, c00);
        c01 = wmma_bf(A0, B1, c01);
        c10 = wmma_bf(A1, B0, c10);
        c11 = wmma_bf(A1, B1, c11);
    }
    v8f cc[2][2] = {{c00, c01}, {c10, c11}};
#pragma unroll
    for (int rt = 0; rt < 2; ++rt) {
#pragma unroll
        for (int ct = 0; ct < 2; ++ct) {
            int o = ocp * 32 + ct * 16 + ln;
            float bias = bo[o];
#pragma unroll
            for (int j = 0; j < 8; ++j)
                out[(size_t)(rowbase + rt * 16 + j + 8 * half) * En + o] = cc[rt][ct][j] + bias;
        }
    }
}

// ---------------------------------------------------------------------------
extern "C" void kernel_launch(void* const* d_in, const int* in_sizes, int n_in,
                              void* d_out, int out_size, void* d_ws, size_t ws_size,
                              hipStream_t stream) {
    const float* xs = (const float*)d_in[0];
    const float* Wq = (const float*)d_in[1];
    const float* bq = (const float*)d_in[2];
    const float* Wk = (const float*)d_in[3];
    const float* bk = (const float*)d_in[4];
    const float* Wv = (const float*)d_in[5];
    const float* bv = (const float*)d_in[6];
    const float* Wo = (const float*)d_in[7];
    const float* bo = (const float*)d_in[8];

    char* ws = (char*)d_ws;
    unsigned short* Wt  = (unsigned short*)ws; ws += (size_t)3 * Hn * DHn * En * 2;   // 6 MB
    unsigned short* WoT = (unsigned short*)ws; ws += (size_t)En * En * 2;             // 2 MB
    unsigned short* Xbf = (unsigned short*)ws; ws += (size_t)Bn * Sn * En * 2;        // 16 MB
    unsigned short* Qbf = (unsigned short*)ws; ws += (size_t)Bn * Hn * Sn * DHn * 2;  // 16 MB
    unsigned short* Kbf = (unsigned short*)ws; ws += (size_t)Bn * Hn * Sn * DHn * 2;  // 16 MB
    unsigned short* Vt  = (unsigned short*)ws; ws += (size_t)Bn * Hn * DHn * Sn * 2;  // 16 MB
    unsigned short* Ctx = (unsigned short*)ws; ws += (size_t)Bn * Sn * En * 2;        // 16 MB

    // 4*2048*1024/8 = 1,048,576 threads
    xs2bf<<<4096, 256, 0, stream>>>(xs, (unsigned int*)Xbf);

    int prepN = 3 * Hn * DHn * En + En * En;                         // 4,194,304
    prep_weights<<<(prepN + 255) / 256, 256, 0, stream>>>(Wq, Wk, Wv, Wo, Wt, WoT);

    // 256 * 2 * 16 * 3 = 24576 waves -> 3072 blocks of 8 waves
    qkv_proj<<<3072, 256, 0, stream>>>(Xbf, Wt, bq, bk, bv, Qbf, Kbf, Vt);

    // 4*16*64 = 4096 waves -> 512 blocks of 8 waves (one (b,h) per block)
    attn<<<512, 256, 0, stream>>>(Qbf, Kbf, Vt, Ctx);

    // 256*32 = 8192 waves -> 1024 blocks
    out_proj<<<1024, 256, 0, stream>>>(Ctx, WoT, bo, (float*)d_out);
}
